// SpcLSTM_48266842472494
// MI455X (gfx1250) — matrix operand order repaired and checked by reading, not compiled, not run on hardware
//
#include <hip/hip_runtime.h>
#include <hip/hip_bf16.h>

typedef _Float16 v8h  __attribute__((ext_vector_type(8)));
typedef _Float16 v16h __attribute__((ext_vector_type(16)));
typedef float    v8f  __attribute__((ext_vector_type(8)));

#define WMMA_F16(A, B, C) \
    __builtin_amdgcn_wmma_f32_16x16x32_f16(false, (A), false, (B), (short)0, (C), false, false)

// Problem constants (from reference)
constexpr int BATCH = 4096;
constexpr int T_ENC = 200;
constexpr int S_DEC = 20;
constexpr int NF    = 16;
constexpr int NT    = 2;
constexpr int H1    = 128;   // 4*H1 = 512 gates
constexpr int H2    = 64;    // 4*H2 = 256 gates
constexpr int H3    = 32;
constexpr int MROWS = 16;    // batch rows per workgroup (one WMMA M-tile)

// Padded row strides (in f16 elements). 272B / 144B / 48B rows break the
// 64-bank (256B) stride pathology: every WMMA fragment load now runs at the
// LDS floor rate (2 cycles per 32-lane 16B/lane access) instead of ~16-way
// conflicts. Pad columns are never read.
constexpr int S1H  = 136;    // Whh1 rows (128 + 8)
constexpr int S2X  = 136;    // Wih2 rows (128 + 8)
constexpr int S2H  = 72;     // Whh2 rows (64 + 8)
constexpr int SAX  = 24;     // x rows (16 + 8)
constexpr int SAH1 = 136;    // h1 rows
constexpr int SAH2 = 72;     // h2 rows

// ---- LDS layout (bytes). All offsets multiples of 32. ----
constexpr int OFF_W1H = 0;                          // Whh1 f16 [512][S1H]
constexpr int OFF_W1X = OFF_W1H + 512*S1H*2;        // Wih1 f16 [512][16] (unpadded)
constexpr int OFF_W2X = OFF_W1X + 512*16*2;         // Wih2 f16 [256][S2X]
constexpr int OFF_W2H = OFF_W2X + 256*S2X*2;        // Whh2 f16 [256][S2H]
constexpr int OFF_WD  = OFF_W2H + 256*S2H*2;        // Wd   f16 [32][64] (scalar use)
constexpr int OFF_AX  = OFF_WD  + 32*64*2;          // x    f16 [16][SAX]
constexpr int OFF_AH1 = OFF_AX  + 16*SAX*2;         // h1   f16 [16][SAH1]
constexpr int OFF_AH2 = OFF_AH1 + 16*SAH1*2;        // h2   f16 [16][SAH2]
constexpr int OFF_G   = OFF_AH2 + 16*SAH2*2;        // gate buf f32 [16][512]
constexpr int OFF_C1  = OFF_G   + 16*512*4;         // c1 f32 [16][128]
constexpr int OFF_C2  = OFF_C1  + 16*128*4;         // c2 f32 [16][64]
constexpr int OFF_B1  = OFF_C2  + 16*64*4;          // bih1+bhh1 f32 [512]
constexpr int OFF_B2  = OFF_B1  + 512*4;            // bih2+bhh2 f32 [256]
constexpr int OFF_BD  = OFF_B2  + 256*4;            // bd f32 [32]
constexpr int OFF_WF  = OFF_BD  + 32*4;             // Wf f32 [2][32]
constexpr int OFF_BF  = OFF_WF  + 64*4;             // bf+out_bias f32 [2] (padded)
constexpr int OFF_D   = OFF_BF  + 32;               // dense act f32 [16][32]
constexpr int SMEM_BYTES = OFF_D + 16*32*4;         // = 324,256 B  (< 320 KB/WGP)

static_assert(SMEM_BYTES <= 320*1024, "LDS budget exceeded");

static __device__ __forceinline__ v8h zero8h() {
    v8h z;
#pragma unroll
    for (int i = 0; i < 8; ++i) z[i] = (_Float16)0;
    return z;
}
static __device__ __forceinline__ v16h zero16h() {
    v16h z;
#pragma unroll
    for (int i = 0; i < 16; ++i) z[i] = (_Float16)0;
    return z;
}
static __device__ __forceinline__ v8f zero8f() {
    v8f z;
#pragma unroll
    for (int i = 0; i < 8; ++i) z[i] = 0.0f;
    return z;
}
static __device__ __forceinline__ v16h cat8(v8h lo, v8h hi) {
    return __builtin_shufflevector(lo, hi, 0,1,2,3,4,5,6,7,8,9,10,11,12,13,14,15);
}
static __device__ __forceinline__ v16h ldfrag(const _Float16* p) {
    // 16 contiguous f16 at 16B alignment -> two aligned v8h loads
    v8h lo = *reinterpret_cast<const v8h*>(p);
    v8h hi = *reinterpret_cast<const v8h*>(p + 8);
    return cat8(lo, hi);
}
static __device__ __forceinline__ float sigm_f(float x) {
    return 1.0f / (1.0f + __expf(-x));          // x->-inf : exp->inf : 0  (no NaN)
}
static __device__ __forceinline__ float tanh_f(float x) {
    float e = __expf(2.0f * x);                 // overflow-safe: e=inf -> 1-0 = 1
    return 1.0f - 2.0f / (e + 1.0f);
}

__global__ __launch_bounds__(256, 1)
void SpcLSTM_48266842472494_kernel(
    const float* __restrict__ xh,   const float* __restrict__ xf,
    const float* __restrict__ Wih1, const float* __restrict__ Whh1,
    const float* __restrict__ bih1, const float* __restrict__ bhh1,
    const float* __restrict__ Wih2, const float* __restrict__ Whh2,
    const float* __restrict__ bih2, const float* __restrict__ bhh2,
    const float* __restrict__ Wdg,  const float* __restrict__ bdg,
    const float* __restrict__ Wfg,  const float* __restrict__ bfg,
    const float* __restrict__ outb, float* __restrict__ out)
{
    extern __shared__ char smem[];
    _Float16* sW1h = reinterpret_cast<_Float16*>(smem + OFF_W1H);
    _Float16* sW1x = reinterpret_cast<_Float16*>(smem + OFF_W1X);
    _Float16* sW2x = reinterpret_cast<_Float16*>(smem + OFF_W2X);
    _Float16* sW2h = reinterpret_cast<_Float16*>(smem + OFF_W2H);
    _Float16* sWd  = reinterpret_cast<_Float16*>(smem + OFF_WD);
    _Float16* sAx  = reinterpret_cast<_Float16*>(smem + OFF_AX);
    _Float16* sAh1 = reinterpret_cast<_Float16*>(smem + OFF_AH1);
    _Float16* sAh2 = reinterpret_cast<_Float16*>(smem + OFF_AH2);
    float*    sG   = reinterpret_cast<float*>(smem + OFF_G);
    float*    sC1  = reinterpret_cast<float*>(smem + OFF_C1);
    float*    sC2  = reinterpret_cast<float*>(smem + OFF_C2);
    float*    sB1  = reinterpret_cast<float*>(smem + OFF_B1);
    float*    sB2  = reinterpret_cast<float*>(smem + OFF_B2);
    float*    sBd  = reinterpret_cast<float*>(smem + OFF_BD);
    float*    sWf  = reinterpret_cast<float*>(smem + OFF_WF);
    float*    sBf  = reinterpret_cast<float*>(smem + OFF_BF);
    float*    sD   = reinterpret_cast<float*>(smem + OFF_D);

    const int tid  = threadIdx.x;
    const int lane = tid & 31;
    const int wv   = tid >> 5;         // 8 waves
    const int lm   = lane & 15;        // N index / M row within half
    const int lh   = lane >> 4;        // lane half select
    const int b0   = blockIdx.x * MROWS;

    // ---------------- one-time: weights -> f16 LDS (padded rows), state init --
    for (int i = tid; i < 512*128; i += 256) {
        const int r = i >> 7, c = i & 127;
        sW1h[r*S1H + c] = (_Float16)Whh1[i];
    }
    for (int i = tid; i < 512*16;  i += 256) sW1x[i] = (_Float16)Wih1[i];
    for (int i = tid; i < 256*128; i += 256) {
        const int r = i >> 7, c = i & 127;
        sW2x[r*S2X + c] = (_Float16)Wih2[i];
    }
    for (int i = tid; i < 256*64;  i += 256) {
        const int r = i >> 6, c = i & 63;
        sW2h[r*S2H + c] = (_Float16)Whh2[i];
    }
    for (int i = tid; i < 32*64;   i += 256) sWd[i]  = (_Float16)Wdg[i];
    for (int i = tid; i < 512;     i += 256) sB1[i]  = bih1[i] + bhh1[i];
    if (tid < 256) sB2[tid] = bih2[tid] + bhh2[tid];
    if (tid < 32)  sBd[tid] = bdg[tid];
    if (tid < 64)  sWf[tid] = Wfg[tid];
    if (tid < 2)   sBf[tid] = bfg[tid] + outb[tid];
    for (int i = tid; i < 16*128; i += 256) {
        const int r = i >> 7, c = i & 127;
        sC1[i] = 0.0f; sAh1[r*SAH1 + c] = (_Float16)0;
    }
    for (int i = tid; i < 16*64; i += 256) {
        const int r = i >> 6, c = i & 63;
        sC2[i] = 0.0f; sAh2[r*SAH2 + c] = (_Float16)0;
    }
    __syncthreads();

    // ---------------- shared two-layer LSTM step (reads sAx, updates h/c) ----
    auto rnn_step = [&]() {
        // ===== Layer 1:  g[16,512] = [x(16) | h1(128)] @ W1^T  =====
        v8f acc[4];
#pragma unroll
        for (int j = 0; j < 4; ++j) acc[j] = zero8f();

        { // x K-tile: real K=0..15, K=16..31 zero-padded on both A and B
            v8h lo = *reinterpret_cast<const v8h*>(sAx + lm*SAX + lh*8);
            v16h a = cat8(lo, zero8h());
#pragma unroll
            for (int j = 0; j < 4; ++j) {
                const int nrow = (wv*4 + j)*16 + lm;
                v16h b = cat8(*reinterpret_cast<const v8h*>(sW1x + nrow*16 + lh*8), zero8h());
                acc[j] = WMMA_F16(a, b, acc[j]);
            }
        }
#pragma unroll
        for (int kt = 0; kt < 4; ++kt) {  // h1 K-tiles (K=128)
            v8h lo = *reinterpret_cast<const v8h*>(sAh1 + lm*SAH1 + kt*32 + lh*16);
            v8h hi = *reinterpret_cast<const v8h*>(sAh1 + lm*SAH1 + kt*32 + lh*16 + 8);
            v16h a = cat8(lo, hi);
#pragma unroll
            for (int j = 0; j < 4; ++j) {
                const int nrow = (wv*4 + j)*16 + lm;
                v16h b = ldfrag(sW1h + nrow*S1H + kt*32 + lh*16);
                acc[j] = WMMA_F16(a, b, acc[j]);
            }
        }
#pragma unroll
        for (int j = 0; j < 4; ++j)
#pragma unroll
            for (int r = 0; r < 8; ++r)
                sG[(lh*8 + r)*512 + (wv*4 + j)*16 + lm] = acc[j][r];
        __syncthreads();

        // LSTM cell 1: 16 x 128 cells
#pragma unroll
        for (int q = 0; q < 8; ++q) {
            const int idx = tid + q*256;
            const int row = idx >> 7, j = idx & 127;
            const float gi = sG[row*512 + j      ] + sB1[j      ];
            const float gf = sG[row*512 + j + 128] + sB1[j + 128];
            const float gg = sG[row*512 + j + 256] + sB1[j + 256];
            const float go = sG[row*512 + j + 384] + sB1[j + 384];
            const float c  = sigm_f(gf) * sC1[row*128 + j] + sigm_f(gi) * tanh_f(gg);
            sC1[row*128 + j]     = c;
            sAh1[row*SAH1 + j]   = (_Float16)(sigm_f(go) * tanh_f(c));
        }
        __syncthreads();

        // ===== Layer 2:  g[16,256] = [h1(128) | h2(64)] @ W2^T  =====
        v8f a2[2];
#pragma unroll
        for (int j = 0; j < 2; ++j) a2[j] = zero8f();
#pragma unroll
        for (int kt = 0; kt < 4; ++kt) {  // h1 K-tiles
            v8h lo = *reinterpret_cast<const v8h*>(sAh1 + lm*SAH1 + kt*32 + lh*16);
            v8h hi = *reinterpret_cast<const v8h*>(sAh1 + lm*SAH1 + kt*32 + lh*16 + 8);
            v16h a = cat8(lo, hi);
#pragma unroll
            for (int j = 0; j < 2; ++j) {
                const int nrow = (wv*2 + j)*16 + lm;
                v16h b = ldfrag(sW2x + nrow*S2X + kt*32 + lh*16);
                a2[j] = WMMA_F16(a, b, a2[j]);
            }
        }
#pragma unroll
        for (int kt = 0; kt < 2; ++kt) {  // h2 K-tiles
            v8h lo = *reinterpret_cast<const v8h*>(sAh2 + lm*SAH2 + kt*32 + lh*16);
            v8h hi = *reinterpret_cast<const v8h*>(sAh2 + lm*SAH2 + kt*32 + lh*16 + 8);
            v16h a = cat8(lo, hi);
#pragma unroll
            for (int j = 0; j < 2; ++j) {
                const int nrow = (wv*2 + j)*16 + lm;
                v16h b = ldfrag(sW2h + nrow*S2H + kt*32 + lh*16);
                a2[j] = WMMA_F16(a, b, a2[j]);
            }
        }
#pragma unroll
        for (int j = 0; j < 2; ++j)
#pragma unroll
            for (int r = 0; r < 8; ++r)
                sG[(lh*8 + r)*512 + (wv*2 + j)*16 + lm] = a2[j][r];
        __syncthreads();

        // LSTM cell 2: 16 x 64 cells
#pragma unroll
        for (int q = 0; q < 4; ++q) {
            const int idx = tid + q*256;
            const int row = idx >> 6, j = idx & 63;
            const float gi = sG[row*512 + j      ] + sB2[j      ];
            const float gf = sG[row*512 + j +  64] + sB2[j +  64];
            const float gg = sG[row*512 + j + 128] + sB2[j + 128];
            const float go = sG[row*512 + j + 192] + sB2[j + 192];
            const float c  = sigm_f(gf) * sC2[row*64 + j] + sigm_f(gi) * tanh_f(gg);
            sC2[row*64 + j]     = c;
            sAh2[row*SAH2 + j]  = (_Float16)(sigm_f(go) * tanh_f(c));
        }
        __syncthreads();
    };

    // ---------------- encoder: 200 steps ----------------
    for (int t = 0; t < T_ENC; ++t) {
        {   // load x_t tile (16 rows x 16 features); prefetch next step
            const int row = tid >> 4, col = tid & 15;
            const size_t base = ((size_t)(b0 + row) * T_ENC + t) * NF + col;
            sAx[row*SAX + col] = (_Float16)xh[base];
            if (t + 1 < T_ENC) __builtin_prefetch(&xh[base + NF], 0, 3);
        }
        __syncthreads();
        rnn_step();
    }
    // sAx cols 14..15 now hold x_history[:, T-1, 14:16] == initial `cur`

    // ---------------- decoder: 20 steps + dense head ----------------
    for (int s = 0; s < S_DEC; ++s) {
        if (tid < 16 * (NF - NT)) {   // ref -> sAx cols 0..13 (cur stays in 14..15)
            const int row = tid / (NF - NT), col = tid % (NF - NT);
            sAx[row*SAX + col] =
                (_Float16)xf[((size_t)(b0 + row) * S_DEC + s) * (NF - NT) + col];
        }
        __syncthreads();
        rnn_step();

        // d = relu(h2 @ Wd^T + bd)  : 16 x 32, K = 64
#pragma unroll
        for (int q = 0; q < 2; ++q) {
            const int idx = tid + q*256;
            const int row = idx >> 5, col = idx & 31;
            float sum = sBd[col];
#pragma unroll
            for (int k = 0; k < H2; ++k)
                sum += (float)sAh2[row*SAH2 + k] * (float)sWd[col*64 + k];
            sD[row*32 + col] = fmaxf(sum, 0.0f);
        }
        __syncthreads();

        // pred = d @ Wf^T + bf + out_bias : 16 x 2, K = 32
        if (tid < 32) {
            const int row = tid >> 1, k = tid & 1;
            float sum = sBf[k];
#pragma unroll
            for (int j = 0; j < H3; ++j)
                sum += sD[row*32 + j] * sWf[k*32 + j];
            out[((size_t)(b0 + row) * S_DEC + s) * NT + k] = sum;
            sAx[row*SAX + (NF - NT) + k] = (_Float16)sum;   // becomes next `cur`
        }
        __syncthreads();
    }
}

extern "C" void kernel_launch(void* const* d_in, const int* in_sizes, int n_in,
                              void* d_out, int out_size, void* d_ws, size_t ws_size,
                              hipStream_t stream) {
    const float* xh   = (const float*)d_in[0];
    const float* xf   = (const float*)d_in[1];
    const float* Wih1 = (const float*)d_in[2];
    const float* Whh1 = (const float*)d_in[3];
    const float* bih1 = (const float*)d_in[4];
    const float* bhh1 = (const float*)d_in[5];
    const float* Wih2 = (const float*)d_in[6];
    const float* Whh2 = (const float*)d_in[7];
    const float* bih2 = (const float*)d_in[8];
    const float* bhh2 = (const float*)d_in[9];
    const float* Wd   = (const float*)d_in[10];
    const float* bd   = (const float*)d_in[11];
    const float* Wf   = (const float*)d_in[12];
    const float* bf   = (const float*)d_in[13];
    const float* ob   = (const float*)d_in[14];
    float* out = (float*)d_out;

    (void)in_sizes; (void)n_in; (void)out_size; (void)d_ws; (void)ws_size;

    hipFuncSetAttribute((const void*)SpcLSTM_48266842472494_kernel,
                        hipFuncAttributeMaxDynamicSharedMemorySize, SMEM_BYTES);

    SpcLSTM_48266842472494_kernel<<<dim3(BATCH / MROWS), dim3(256), SMEM_BYTES, stream>>>(
        xh, xf, Wih1, Whh1, bih1, bhh1, Wih2, Whh2, bih2, bhh2,
        Wd, bd, Wf, bf, ob, out);
}
